// EmergentVision_19353122636153
// MI455X (gfx1250) — matrix operand order repaired and checked by reading, not compiled, not run on hardware
//
#include <hip/hip_runtime.h>
#include <math.h>
#include <stdint.h>

// ---------------------------------------------------------------------------
// Types for CDNA5 WMMA (wave32): v_wmma_f32_16x16x32_bf16
// ---------------------------------------------------------------------------
typedef __bf16 bf16_t;
typedef __attribute__((ext_vector_type(16))) __bf16 bfx16;
typedef __attribute__((ext_vector_type(8)))  __bf16 bfx8;
typedef __attribute__((ext_vector_type(8)))  float  fx8;

#define EPSV  1e-5f
#define LEAKV 0.95f

static __device__ __forceinline__ bf16_t to_bf16(float f) {
    union { float f; unsigned u; } uf; uf.f = f;
    unsigned r = uf.u + 0x7FFFu + ((uf.u >> 16) & 1u);   // round-to-nearest-even
    unsigned short hs = (unsigned short)(r >> 16);
    union { unsigned short s; bf16_t b; } ub; ub.s = hs;
    return ub.b;
}

// ---------------------------------------------------------------------------
// fp32 -> bf16 pack, optional magnitude mask (|w| > thr else 0)
// ---------------------------------------------------------------------------
__global__ __launch_bounds__(256)
void pack_bf16_k(const float* __restrict__ w, bf16_t* __restrict__ o,
                 size_t n, float mthr, int useMask)
{
    size_t i = (size_t)blockIdx.x * 256 + threadIdx.x;
    size_t stride = (size_t)gridDim.x * 256;
    for (; i < n; i += stride) {
        float v = w[i];
        if (useMask && !(fabsf(v) > mthr)) v = 0.0f;
        o[i] = to_bf16(v);
    }
}

// ---------------------------------------------------------------------------
// conv1 (1->32, 3x3 SAME, 28x28) + BN + ReLU + 2x2 maxpool -> f32 [256,32,14,14]
// ---------------------------------------------------------------------------
__global__ __launch_bounds__(256)
void conv1_fused_k(const float* __restrict__ x, const float* __restrict__ w,
                   const float* __restrict__ cb,
                   const float* __restrict__ g,  const float* __restrict__ bb,
                   const float* __restrict__ bm, const float* __restrict__ bv,
                   float* __restrict__ out)
{
    int idx = blockIdx.x * 256 + threadIdx.x;
    if (idx >= 256 * 32 * 14 * 14) return;
    int ox = idx % 14, oy = (idx / 14) % 14, oc = (idx / 196) % 32, b = idx / (196 * 32);
    float inv = g[oc] * rsqrtf(bv[oc] + EPSV);
    float sh  = bb[oc] - bm[oc] * inv;
    const float* xi = x + (size_t)b * 784;
    const float* wk = w + oc * 9;
    float best = 0.0f;                       // relu outputs are >= 0
    for (int py = 0; py < 2; ++py)
    for (int px = 0; px < 2; ++px) {
        int y = oy * 2 + py, xx = ox * 2 + px;
        float s = cb[oc];
        #pragma unroll
        for (int ky = 0; ky < 3; ++ky) {
            int iy = y + ky - 1;
            if (iy < 0 || iy >= 28) continue;
            #pragma unroll
            for (int kx = 0; kx < 3; ++kx) {
                int ix = xx + kx - 1;
                if (ix < 0 || ix >= 28) continue;
                s += wk[ky * 3 + kx] * xi[iy * 28 + ix];
            }
        }
        float v = s * inv + sh;
        v = v > 0.0f ? v : 0.0f;
        best = v > best ? v : best;
    }
    out[idx] = best;
}

// ---------------------------------------------------------------------------
// conv2 (32->64, 3x3 SAME, 14x14) + BN + ReLU + 2x2 maxpool
// -> bf16 [256, 3136] flattened (c*49 + oy*7 + ox), ready as WMMA A matrix
// ---------------------------------------------------------------------------
__global__ __launch_bounds__(256)
void conv2_fused_k(const float* __restrict__ h1, const float* __restrict__ w,
                   const float* __restrict__ cb,
                   const float* __restrict__ g,  const float* __restrict__ bb,
                   const float* __restrict__ bm, const float* __restrict__ bv,
                   bf16_t* __restrict__ out)
{
    int idx = blockIdx.x * 256 + threadIdx.x;
    if (idx >= 256 * 64 * 49) return;
    int ox = idx % 7, oy = (idx / 7) % 7, oc = (idx / 49) % 64, b = idx / (49 * 64);
    float inv = g[oc] * rsqrtf(bv[oc] + EPSV);
    float sh  = bb[oc] - bm[oc] * inv;
    const float* hb = h1 + (size_t)b * 32 * 196;
    const float* wk = w + (size_t)oc * 32 * 9;
    float best = 0.0f;
    for (int py = 0; py < 2; ++py)
    for (int px = 0; px < 2; ++px) {
        int y = oy * 2 + py, xx = ox * 2 + px;
        float s = cb[oc];
        for (int ic = 0; ic < 32; ++ic) {
            const float* hc = hb + ic * 196;
            const float* wc = wk + ic * 9;
            #pragma unroll
            for (int ky = 0; ky < 3; ++ky) {
                int iy = y + ky - 1;
                if (iy < 0 || iy >= 14) continue;
                #pragma unroll
                for (int kx = 0; kx < 3; ++kx) {
                    int ix = xx + kx - 1;
                    if (ix < 0 || ix >= 14) continue;
                    s += wc[ky * 3 + kx] * hc[iy * 14 + ix];
                }
            }
        }
        float v = s * inv + sh;
        v = v > 0.0f ? v : 0.0f;
        best = v > best ? v : best;
    }
    out[(size_t)b * 3136 + oc * 49 + oy * 7 + ox] = to_bf16(best);
}

// ---------------------------------------------------------------------------
// bf16 WMMA GEMM: C[M,N] = A[M,K] * Bt[N,K]^T (+ bias[N])
// Outer-product register tiling: each wave computes a 32(M) x 32(N) tile as
// 2 A fragments x 2 B fragments -> 4 accumulators, with an explicit 2-stage
// software pipeline (loads for k+1 issued before WMMAs of k).
// Block = 4 waves (128 thr) covering 32M x 128N; all waves share the same
// 32 A rows so A loads dedupe in L0. Grid = (N/128, M/32).
// Layouts (wave32, 16-bit A/B, f32 C/D); A and B use the same contiguous k order, not the table order:
//   A: lanes 0-15 (M=l) hold K=k0..k0+15 contiguous (row of A); lanes 16-31 K+16
//   B: lanes 0-15 (N=l) hold K=k0..k0+15 contiguous (row of Bt); lanes 16-31 K+16
//   C: vgpr r holds row M = r (lanes 0-15) / 8+r (lanes 16-31), N = lane&15
// Requires: M%32==0, N%128==0, K%32==0, K>=64 (true for all call sites).
// ---------------------------------------------------------------------------
static __device__ __forceinline__ bfx16 load_a_frag(const bf16_t* p) {
    bfx8 lo = *(const bfx8*)(p);
    bfx8 hi = *(const bfx8*)(p + 8);
    bfx16 a;
    #pragma unroll
    for (int i = 0; i < 8; ++i) { a[i] = lo[i]; a[8 + i] = hi[i]; }
    return a;
}

__global__ __launch_bounds__(128)
void gemm_bf16_wmma_k(const bf16_t* __restrict__ A, const bf16_t* __restrict__ Bt,
                      float* __restrict__ C, const float* __restrict__ bias,
                      int M, int N, int K, int useBias)
{
    const int lane = threadIdx.x & 31;
    const int wave = threadIdx.x >> 5;
    const int half = lane >> 4;     // 0: lanes 0-15, 1: lanes 16-31
    const int l15  = lane & 15;

    const int tm  = blockIdx.y;                   // 32-row M tile
    const int tn0 = blockIdx.x * 8 + wave * 2;    // first of 2 16-col N tiles

    const bf16_t* ap0 = A  + (size_t)(tm * 32 + l15) * K + half * 16;
    const bf16_t* ap1 = ap0 + (size_t)16 * K;
    const bf16_t* bp0 = Bt + (size_t)((tn0 + 0) * 16 + l15) * K + half * 16;
    const bf16_t* bp1 = Bt + (size_t)((tn0 + 1) * 16 + l15) * K + half * 16;

    fx8 acc00 = {}; fx8 acc01 = {}; fx8 acc10 = {}; fx8 acc11 = {};

    // pipeline stage 0
    bfx16 a0c = load_a_frag(ap0);
    bfx16 a1c = load_a_frag(ap1);
    bfx16 b0c = *(const bfx16*)(bp0);
    bfx16 b1c = *(const bfx16*)(bp1);

    for (int k0 = 32; k0 < K; k0 += 32) {
        // issue next chunk's loads before consuming the current fragments
        bfx16 a0n = load_a_frag(ap0 + k0);
        bfx16 a1n = load_a_frag(ap1 + k0);
        bfx16 b0n = *(const bfx16*)(bp0 + k0);
        bfx16 b1n = *(const bfx16*)(bp1 + k0);

        // pull the following chunk of the B stream toward the near cache
        __builtin_prefetch(bp0 + k0 + 32, 0, 3);
        __builtin_prefetch(bp1 + k0 + 32, 0, 3);

        acc00 = __builtin_amdgcn_wmma_f32_16x16x32_bf16(false, a0c, false, b0c, (short)0, acc00, false, false);
        acc01 = __builtin_amdgcn_wmma_f32_16x16x32_bf16(false, a0c, false, b1c, (short)0, acc01, false, false);
        acc10 = __builtin_amdgcn_wmma_f32_16x16x32_bf16(false, a1c, false, b0c, (short)0, acc10, false, false);
        acc11 = __builtin_amdgcn_wmma_f32_16x16x32_bf16(false, a1c, false, b1c, (short)0, acc11, false, false);

        a0c = a0n; a1c = a1n; b0c = b0n; b1c = b1n;
    }

    // pipeline drain
    acc00 = __builtin_amdgcn_wmma_f32_16x16x32_bf16(false, a0c, false, b0c, (short)0, acc00, false, false);
    acc01 = __builtin_amdgcn_wmma_f32_16x16x32_bf16(false, a0c, false, b1c, (short)0, acc01, false, false);
    acc10 = __builtin_amdgcn_wmma_f32_16x16x32_bf16(false, a1c, false, b0c, (short)0, acc10, false, false);
    acc11 = __builtin_amdgcn_wmma_f32_16x16x32_bf16(false, a1c, false, b1c, (short)0, acc11, false, false);

    const int cn0  = (tn0 + 0) * 16 + l15;
    const int cn1  = (tn0 + 1) * 16 + l15;
    const float b0v = useBias ? bias[cn0] : 0.0f;
    const float b1v = useBias ? bias[cn1] : 0.0f;
    const int r0 = tm * 32 + half * 8;        // rows for acc0x
    const int r1 = r0 + 16;                   // rows for acc1x
    #pragma unroll
    for (int r = 0; r < 8; ++r) {
        C[(size_t)(r0 + r) * N + cn0] = acc00[r] + b0v;
        C[(size_t)(r0 + r) * N + cn1] = acc01[r] + b1v;
        C[(size_t)(r1 + r) * N + cn0] = acc10[r] + b0v;
        C[(size_t)(r1 + r) * N + cn1] = acc11[r] + b1v;
    }
}

// ---------------------------------------------------------------------------
// Fused layernorm + PulseTGAU neuron step over one [4096] row per block.
// Emits fp32 spikes (for classifier) and bf16 spikes (next GEMM A operand).
// ---------------------------------------------------------------------------
__global__ __launch_bounds__(256)
void ln_neuron_k(const float* __restrict__ xin, float* __restrict__ pot,
                 float* __restrict__ spikes, bf16_t* __restrict__ spikes_bf,
                 const float* __restrict__ lg, const float* __restrict__ lb,
                 const float* __restrict__ thr, const float* __restrict__ intr,
                 int firstStep)
{
    __shared__ float s1[256], s2[256];
    const int row = blockIdx.x;
    const int tid = threadIdx.x;
    const size_t base = (size_t)row * 4096;

    float v[16];
    float a1 = 0.0f, a2 = 0.0f;
    #pragma unroll
    for (int i = 0; i < 16; ++i) {
        float t = xin[base + i * 256 + tid];
        v[i] = t; a1 += t; a2 += t * t;
    }
    s1[tid] = a1; s2[tid] = a2; __syncthreads();
    for (int off = 128; off > 0; off >>= 1) {
        if (tid < off) { s1[tid] += s1[tid + off]; s2[tid] += s2[tid + off]; }
        __syncthreads();
    }
    const float mean = s1[0] * (1.0f / 4096.0f);
    const float var  = s2[0] * (1.0f / 4096.0f) - mean * mean;
    const float rstd = rsqrtf(var + EPSV);

    #pragma unroll
    for (int i = 0; i < 16; ++i) {
        const int col = i * 256 + tid;
        float t  = (v[i] - mean) * rstd * lg[col] + lb[col];
        float p0 = firstStep ? 0.0f : pot[base + col];
        float p  = p0 * LEAKV + t + intr[col];
        float gg = p - thr[col];
        float sp = 0.0f;
        if (gg > 0.0f) {                 // mask * silu(gated); reset potential
            sp = gg / (1.0f + __expf(-gg));
            p  = p - gg;                 // == threshold
        }
        pot[base + col]       = p;
        spikes[base + col]    = sp;
        spikes_bf[base + col] = to_bf16(sp);
    }
}

// ---------------------------------------------------------------------------
// Classifier: out[b,c] = dot(spikes[b,:], cls_w[c,:]) + cls_b[c]
// ---------------------------------------------------------------------------
__global__ __launch_bounds__(256)
void classifier_k(const float* __restrict__ spikes, const float* __restrict__ cw,
                  const float* __restrict__ cb, float* __restrict__ out)
{
    __shared__ float sm[256];
    const int c = blockIdx.x % 10;
    const int b = blockIdx.x / 10;
    const int tid = threadIdx.x;
    const float* sr = spikes + (size_t)b * 4096;
    const float* wr = cw + (size_t)c * 4096;
    float a = 0.0f;
    for (int i = tid; i < 4096; i += 256) a += sr[i] * wr[i];
    sm[tid] = a; __syncthreads();
    for (int off = 128; off > 0; off >>= 1) {
        if (tid < off) sm[tid] += sm[tid + off];
        __syncthreads();
    }
    if (tid == 0) out[b * 10 + c] = sm[0] + cb[c];
}

// ---------------------------------------------------------------------------
// Launch
// ---------------------------------------------------------------------------
extern "C" void kernel_launch(void* const* d_in, const int* in_sizes, int n_in,
                              void* d_out, int out_size, void* d_ws, size_t ws_size,
                              hipStream_t stream)
{
    (void)in_sizes; (void)n_in; (void)out_size; (void)ws_size;

    const float* x     = (const float*)d_in[0];
    const float* c1w   = (const float*)d_in[1];
    const float* c1b   = (const float*)d_in[2];
    const float* bn1g  = (const float*)d_in[3];
    const float* bn1b  = (const float*)d_in[4];
    const float* bn1m  = (const float*)d_in[5];
    const float* bn1v  = (const float*)d_in[6];
    const float* c2w   = (const float*)d_in[7];
    const float* c2b   = (const float*)d_in[8];
    const float* bn2g  = (const float*)d_in[9];
    const float* bn2b  = (const float*)d_in[10];
    const float* bn2m  = (const float*)d_in[11];
    const float* bn2v  = (const float*)d_in[12];
    const float* fcw   = (const float*)d_in[13];
    const float* fcb   = (const float*)d_in[14];
    const float* recw  = (const float*)d_in[15];
    const float* thr   = (const float*)d_in[16];
    const float* intr  = (const float*)d_in[17];
    const float* lng   = (const float*)d_in[18];
    const float* lnb   = (const float*)d_in[19];
    const float* clsw  = (const float*)d_in[20];
    const float* clsb  = (const float*)d_in[21];

    // workspace carve (256B aligned)
    uint8_t* ws = (uint8_t*)d_ws;
    size_t off = 0;
    auto carve = [&](size_t bytes) -> void* {
        void* p = ws + off;
        off += (bytes + 255) & ~(size_t)255;
        return p;
    };
    bf16_t* effw  = (bf16_t*)carve((size_t)4096 * 4096 * 2); // masked rec_w, bf16 (32MB, L2-resident)
    bf16_t* fcwb  = (bf16_t*)carve((size_t)4096 * 3136 * 2); // fc_w bf16
    float*  h1    = (float*) carve((size_t)256 * 32 * 196 * 4);
    bf16_t* h2    = (bf16_t*)carve((size_t)256 * 3136 * 2);
    float*  cur   = (float*) carve((size_t)256 * 4096 * 4);  // initial_current
    float*  tot   = (float*) carve((size_t)256 * 4096 * 4);  // step GEMM output
    float*  spk   = (float*) carve((size_t)256 * 4096 * 4);
    bf16_t* spkb  = (bf16_t*)carve((size_t)256 * 4096 * 2);
    float*  potb  = (float*) carve((size_t)256 * 4096 * 4);

    // weight packing (deterministic, every call)
    pack_bf16_k<<<4096, 256, 0, stream>>>(recw, effw, (size_t)4096 * 4096, 0.01f, 1);
    pack_bf16_k<<<4096, 256, 0, stream>>>(fcw,  fcwb, (size_t)4096 * 3136, 0.0f,  0);

    // vision frontend
    conv1_fused_k<<<(256 * 32 * 196) / 256, 256, 0, stream>>>(x, c1w, c1b, bn1g, bn1b, bn1m, bn1v, h1);
    conv2_fused_k<<<(256 * 64 * 49) / 256, 256, 0, stream>>>(h1, c2w, c2b, bn2g, bn2b, bn2m, bn2v, h2);

    // GEMM grid: (N/128, M/32) with 128 threads (4 waves) per block
    dim3 ggrid(4096 / 128, 256 / 32);   // (32, 8)

    // initial_current = h @ fc_w^T + fc_b   (M=256, N=4096, K=3136)
    gemm_bf16_wmma_k<<<ggrid, 128, 0, stream>>>(h2, fcwb, cur, fcb, 256, 4096, 3136, 1);

    // step 0: spikes==0 -> total = initial_current
    ln_neuron_k<<<256, 256, 0, stream>>>(cur, potb, spk, spkb, lng, lnb, thr, intr, 1);

    // steps 1..11: total = spikes @ eff_w^T  (M=256, N=4096, K=4096)
    for (int s = 1; s < 12; ++s) {
        gemm_bf16_wmma_k<<<ggrid, 128, 0, stream>>>(spkb, effw, tot, fcb, 256, 4096, 4096, 0);
        ln_neuron_k<<<256, 256, 0, stream>>>(tot, potb, spk, spkb, lng, lnb, thr, intr, 0);
    }

    // logits
    classifier_k<<<2560, 256, 0, stream>>>(spk, clsw, clsb, (float*)d_out);
}
